// FEDformer_32057635897487
// MI455X (gfx1250) — matrix operand order repaired and checked by reading, not compiled, not run on hardware
//
#include <hip/hip_runtime.h>

// ---------------------------------------------------------------------------
// FEDformer forward for MI455X (gfx1250, wave32, WMMA).
// Heavy contractions run on v_wmma_f32_16x16x32_bf16 (f32 accumulate),
// activations/weights bf16 end-to-end. GEMM staging uses CDNA5 async
// global->LDS DMA (ASYNCcnt) with LDS double-buffering: tile k+1 is fetched
// behind the WMMAs of tile k via a partial s_wait_asynccnt 6 (async ops
// complete in order per wave). Last tile is peeled so the hot loop is
// branch-free. rFFT/irFFT are dense GEMMs against precomputed cos/sin tables
// (only MODES=64 of 257 frequencies survive).
// ---------------------------------------------------------------------------

typedef __attribute__((ext_vector_type(16))) __bf16 v16bf;
typedef __attribute__((ext_vector_type(8)))  float  v8f;
typedef __attribute__((ext_vector_type(4)))  int    v4i;

union Frag16 { v16bf v; unsigned u[8]; };

#if defined(__has_builtin)
#if __has_builtin(__builtin_amdgcn_global_load_async_to_lds_b128)
#define HAVE_ASYNC_LDS 1
#endif
#endif

#if defined(HAVE_ASYNC_LDS)
typedef __attribute__((address_space(1))) v4i* gv4i_p;   // global v4i*
typedef __attribute__((address_space(3))) v4i* lv4i_p;   // LDS v4i*
template<int N>
__device__ __forceinline__ void wait_async() {
#if __has_builtin(__builtin_amdgcn_s_wait_asynccnt)
  __builtin_amdgcn_s_wait_asynccnt(N);
#else
  asm volatile("s_wait_asynccnt %0" :: "i"(N) : "memory");
#endif
}
#endif

__device__ __forceinline__ unsigned short f2bfu(float f) {
  union { float f; unsigned u; } a; a.f = f;
  unsigned r = a.u + 0x7FFFu + ((a.u >> 16) & 1u);   // round-to-nearest-even
  return (unsigned short)(r >> 16);
}
__device__ __forceinline__ float bfu2f(unsigned short s) {
  union { unsigned u; float f; } a; a.u = ((unsigned)s) << 16; return a.f;
}

// ---------------------------------------------------------------------------
// WMMA GEMM: C(MxN) = op( A(MxK) @ B^T )  [+bias] [+res] [relu]
// A, B, Res, C bf16 row-major; B stored (N,K); bias f32.
// Workgroup tile 128x64, 8 waves, wave tile 32x32 (2x2 WMMA 16x16x32 tiles),
// K staged 64 deep (2 WMMA k-steps), double-buffered in LDS. Pitch 72 shorts
// = 144B keeps rows 16B-aligned (b128) and fragment reads conflict-free.
// Requires M%128==0, N%64==0, K%64==0 and K>=128 (true for all uses here).
// ---------------------------------------------------------------------------
template<bool BIAS, bool RELU, bool RES>
__global__ __launch_bounds__(256)
void k_gemm(const unsigned short* __restrict__ A,
            const unsigned short* __restrict__ Bw,
            const float* __restrict__ bias,
            const unsigned short* __restrict__ Res,
            unsigned short* __restrict__ C, int M, int N, int K) {
  __shared__ __attribute__((aligned(16))) unsigned short As[2 * 128 * 72]; // [buf][m][k]
  __shared__ __attribute__((aligned(16))) unsigned short Bs[2 * 64 * 72];  // [buf][n][k]
  constexpr int ABUF = 128 * 72;
  constexpr int BBUF = 64 * 72;

  const int tid   = threadIdx.x;
  const int lane  = tid & 31;
  const int wave  = tid >> 5;
  const int waveM = (wave & 3) * 32;   // 4 waves along M
  const int waveN = (wave >> 2) * 32;  // 2 waves along N
  const int lhi   = lane >> 4;
  const int llo   = lane & 15;
  const int m0    = blockIdx.y * 128;
  const int n0    = blockIdx.x * 64;

  // Hoisted per-thread staging addresses: idx = tid + j*256 decomposes into
  // row = (tid>>3) + 32j, col = (tid&7)*8 (col is K-tile-invariant).
  const int rb = tid >> 3;
  const int cc = (tid & 7) * 8;
  const unsigned short* pa[4];
  const unsigned short* pb[2];
  int lva[4], lvb[2];
#pragma unroll
  for (int j = 0; j < 4; ++j) {
    pa[j]  = A + (size_t)(m0 + rb + 32 * j) * K + cc;
    lva[j] = (rb + 32 * j) * 72 + cc;
  }
#pragma unroll
  for (int j = 0; j < 2; ++j) {
    pb[j]  = Bw + (size_t)(n0 + rb + 32 * j) * K + cc;
    lvb[j] = (rb + 32 * j) * 72 + cc;
  }

  v8f acc[2][2];
#pragma unroll
  for (int i = 0; i < 2; ++i)
#pragma unroll
    for (int j = 0; j < 2; ++j)
#pragma unroll
      for (int r = 0; r < 8; ++r) acc[i][j][r] = 0.0f;

  // 2 WMMA k-steps on the staged tile at (baseA, baseB)
  auto compute_tile = [&](int baseA, int baseB) {
#pragma unroll
    for (int s = 0; s < 2; ++s) {
      const int kb = s * 32;
      Frag16 fa[2], fb[2];
#pragma unroll
      for (int mt = 0; mt < 2; ++mt) {
        int m = waveM + mt * 16 + llo;                 // A: lane holds row M
#pragma unroll
        for (int v = 0; v < 8; ++v) {
          int ko = lhi * 8 + (v & 3) * 2 + ((v >> 2) << 4);  // K pair per VGPR
          fa[mt].u[v] =
              *reinterpret_cast<const unsigned*>(&As[baseA + m * 72 + kb + ko]);
        }
      }
#pragma unroll
      for (int nt = 0; nt < 2; ++nt) {
        int n  = waveN + nt * 16 + llo;                // B: lane holds col N
        int kg = lhi * 8;                              // K pair per VGPR (same order as A)
#pragma unroll
        for (int v = 0; v < 8; ++v)
          fb[nt].u[v] = *reinterpret_cast<const unsigned*>(
              &Bs[baseB + n * 72 + kb + kg + (v & 3) * 2 + ((v >> 2) << 4)]);
      }
#pragma unroll
      for (int mt = 0; mt < 2; ++mt)
#pragma unroll
        for (int nt = 0; nt < 2; ++nt)
          acc[mt][nt] = __builtin_amdgcn_wmma_f32_16x16x32_bf16(
              false, fa[mt].v, false, fb[nt].v, (short)0, acc[mt][nt], false, false);
    }
  };

#if defined(HAVE_ASYNC_LDS)
  // ---- async DMA double-buffered pipeline (last tile peeled) ----
  auto stage_tile = [&](int baseA, int baseB) {
#pragma unroll
    for (int j = 0; j < 4; ++j) {
      __builtin_amdgcn_global_load_async_to_lds_b128(
          (gv4i_p)pa[j], (lv4i_p)&As[baseA + lva[j]], 0, 0);
      pa[j] += 64;
    }
#pragma unroll
    for (int j = 0; j < 2; ++j) {
      __builtin_amdgcn_global_load_async_to_lds_b128(
          (gv4i_p)pb[j], (lv4i_p)&Bs[baseB + lvb[j]], 0, 0);
      pb[j] += 64;
    }
  };

  const int ntile = K >> 6;
  stage_tile(0, 0);                       // prologue: tile 0 -> buffer 0
  int kt = 0;
#pragma unroll 2
  for (; kt < ntile - 1; ++kt) {          // branch-free steady state
    const int cur = kt & 1;
    stage_tile((cur ^ 1) * ABUF, (cur ^ 1) * BBUF);
    wait_async<6>();                      // oldest 6 (tile kt) done; next in flight
    __syncthreads();                      // all waves' tile-kt data visible
    compute_tile(cur * ABUF, cur * BBUF);
    __syncthreads();                      // protect buffer before next overwrite
  }
  wait_async<0>();                        // tail tile
  __syncthreads();
  compute_tile((kt & 1) * ABUF, (kt & 1) * BBUF);
  // no barrier needed: epilogue touches only registers + global memory
#else
  // ---- fallback: single-buffered, batched b128 load/store ----
  for (int k0 = 0; k0 < K; k0 += 64) {
    uint4 ta[4], tb[2];
#pragma unroll
    for (int j = 0; j < 4; ++j) ta[j] = *reinterpret_cast<const uint4*>(pa[j]);
#pragma unroll
    for (int j = 0; j < 2; ++j) tb[j] = *reinterpret_cast<const uint4*>(pb[j]);
#pragma unroll
    for (int j = 0; j < 4; ++j) {
      *reinterpret_cast<uint4*>(&As[lva[j]]) = ta[j];
      pa[j] += 64;
    }
#pragma unroll
    for (int j = 0; j < 2; ++j) {
      *reinterpret_cast<uint4*>(&Bs[lvb[j]]) = tb[j];
      pb[j] += 64;
    }
    if (k0 + 64 < K) __builtin_prefetch(pa[0], 0, 1);  // global_prefetch_b8
    __syncthreads();
    compute_tile(0, 0);
    __syncthreads();
  }
#endif

  // ---- epilogue (C/D layout: M = lhi*8 + r, N = llo) ----
#pragma unroll
  for (int mt = 0; mt < 2; ++mt) {
#pragma unroll
    for (int nt = 0; nt < 2; ++nt) {
#pragma unroll
      for (int r = 0; r < 8; ++r) {
        int Mi = m0 + waveM + mt * 16 + lhi * 8 + r;
        int Ni = n0 + waveN + nt * 16 + llo;
        float v = acc[mt][nt][r];
        if constexpr (BIAS) v += bias[Ni];
        if constexpr (RES)  v += bfu2f(Res[(size_t)Mi * N + Ni]);
        if constexpr (RELU) v = fmaxf(v, 0.0f);
        C[(size_t)Mi * N + Ni] = f2bfu(v);
      }
    }
  }
}

// ---------------------------------------------------------------------------
// f32 -> bf16 bulk conversion (weights), vectorized 4 per thread.
// ---------------------------------------------------------------------------
__global__ __launch_bounds__(256)
void k_f2bf(const float4* __restrict__ in, ushort4* __restrict__ out, int n4) {
  int i = blockIdx.x * 256 + threadIdx.x;
  if (i < n4) {
    float4 v = in[i];
    ushort4 o;
    o.x = f2bfu(v.x); o.y = f2bfu(v.y); o.z = f2bfu(v.z); o.w = f2bfu(v.w);
    out[i] = o;
  }
}

// ---------------------------------------------------------------------------
// Batched rectangular f32 transpose: out[b][c][r] = in[b][r][c]
// ---------------------------------------------------------------------------
__global__ void k_transpose_f32(const float* __restrict__ in,
                                float* __restrict__ out, int R, int C) {
  __shared__ float tile[32][33];
  const int b  = blockIdx.z;
  const int r0 = blockIdx.y * 32, c0 = blockIdx.x * 32;
  const int tx = threadIdx.x, ty = threadIdx.y;
  const size_t base = (size_t)b * R * C;
  for (int i = ty; i < 32; i += 8) {
    int r = r0 + i, c = c0 + tx;
    if (r < R && c < C) tile[i][tx] = in[base + (size_t)r * C + c];
  }
  __syncthreads();
  for (int i = ty; i < 32; i += 8) {
    int c = c0 + i, r = r0 + tx;
    if (r < R && c < C) out[base + (size_t)c * R + r] = tile[tx][i];
  }
}

// ---------------------------------------------------------------------------
// Batched bf16 transpose with optional fused residual add (f32 math):
//   out[b][c][r] = in[b][r][c] (+ res[b][c][r])
// ---------------------------------------------------------------------------
template<bool ADD>
__global__ void k_transpose_bf(const unsigned short* __restrict__ in,
                               const unsigned short* __restrict__ res,
                               unsigned short* __restrict__ out, int R, int C) {
  __shared__ unsigned short tile[32][34];
  const int b  = blockIdx.z;
  const int r0 = blockIdx.y * 32, c0 = blockIdx.x * 32;
  const int tx = threadIdx.x, ty = threadIdx.y;
  const size_t base = (size_t)b * R * C;
  for (int i = ty; i < 32; i += 8) {
    int r = r0 + i, c = c0 + tx;
    if (r < R && c < C) tile[i][tx] = in[base + (size_t)r * C + c];
  }
  __syncthreads();
  for (int i = ty; i < 32; i += 8) {
    int c = c0 + i, r = r0 + tx;
    if (r < R && c < C) {
      unsigned short v = tile[tx][i];
      if constexpr (ADD)
        v = f2bfu(bfu2f(v) + bfu2f(res[base + (size_t)c * R + r]));
      out[base + (size_t)c * R + r] = v;
    }
  }
}

// ---------------------------------------------------------------------------
// DFT matrices, generated bf16 directly in (N,K) layout for the GEMM:
//  BfT (128,512): row n<64 = cos(2pi n t/512), n>=64 = -sin(2pi (n-64) t/512)
//  BiT (512,128): col r<64 = s_m cos/512, r>=64 = -s_m sin/512 (s_0=1, else 2)
// ---------------------------------------------------------------------------
__global__ void k_init_dft(unsigned short* __restrict__ BfT,
                           unsigned short* __restrict__ BiT) {
  const int idx = blockIdx.x * 256 + threadIdx.x;  // 0..65535
  const float w = 0.01227184630308513f;            // 2*pi/512
  { int n = idx >> 9, t = idx & 511, m = n & 63;
    float ang = w * (float)(m * t);
    BfT[idx] = f2bfu((n < 64) ? cosf(ang) : -sinf(ang)); }
  { int t = idx >> 7, r = idx & 127, m = r & 63;
    float s   = (m == 0) ? 1.0f : 2.0f;
    float ang = w * (float)(m * t);
    float inv = s * (1.0f / 512.0f);
    BiT[idx] = f2bfu((r < 64) ? inv * cosf(ang) : -inv * sinf(ang)); }
}

// ---------------------------------------------------------------------------
// Token embedding + sinusoidal positional encoding -> bf16 activations.
// xT: (B,T,200) f32, Wt: (200,512) f32 pre-transposed W_embed.
// ---------------------------------------------------------------------------
__global__ __launch_bounds__(256)
void k_embed(const float* __restrict__ xT, const float* __restrict__ Wt,
             const float* __restrict__ bemb, unsigned short* __restrict__ out) {
  __shared__ float xs[200];
  const int bt = blockIdx.x;        // b*512 + t
  const int t  = bt & 511;
  const int tid = threadIdx.x;
  if (tid < 200) xs[tid] = xT[(size_t)bt * 200 + tid];
  __syncthreads();
  for (int d = tid; d < 512; d += 256) {
    float acc = bemb[d];
    for (int n = 0; n < 200; ++n) acc = fmaf(xs[n], Wt[n * 512 + d], acc);
    float div = __expf(-0.017988946039015983f * (float)(d & ~1)); // ln(1e4)/512
    float ang = (float)t * div;
    acc += (d & 1) ? __cosf(ang) : __sinf(ang);
    out[(size_t)bt * 512 + d] = f2bfu(acc);
  }
}

// ---------------------------------------------------------------------------
// Complex mode mixing: G[b,h,o,m] = sum_i F[b,h,i,m] * (wre + j*wim)[h,i,o,m]
// F/G bf16, rows (b*512 + h*64 + ch), cols [0..63]=re, [64..127]=im.
// ---------------------------------------------------------------------------
__global__ __launch_bounds__(256)
void k_modemix(const unsigned short* __restrict__ F,
               const float* __restrict__ wre, const float* __restrict__ wim,
               unsigned short* __restrict__ G) {
  const int idx = blockIdx.x * 256 + threadIdx.x;  // < 16384*64
  const int m   = idx & 63;
  const int row = idx >> 6;           // b*512 + h*64 + o
  const int o   = row & 63;
  const int h   = (row >> 6) & 7;
  const int b   = row >> 9;
  const unsigned short* xp = F + ((size_t)(b * 512 + h * 64)) * 128 + m;
  const float* rp = wre + (size_t)h * 262144 + (size_t)o * 64 + m;
  const float* ip = wim + (size_t)h * 262144 + (size_t)o * 64 + m;
  float gr = 0.f, gi = 0.f;
#pragma unroll 4
  for (int i = 0; i < 64; ++i) {
    float ar = bfu2f(xp[(size_t)i * 128]);
    float ai = bfu2f(xp[(size_t)i * 128 + 64]);
    float br = rp[(size_t)i * 4096];
    float bi = ip[(size_t)i * 4096];
    gr = fmaf(ar, br, gr); gr = fmaf(-ai, bi, gr);
    gi = fmaf(ar, bi, gi); gi = fmaf(ai, br, gi);
  }
  G[(size_t)row * 128 + m]      = f2bfu(gr);
  G[(size_t)row * 128 + 64 + m] = f2bfu(gi);
}

// ---------------------------------------------------------------------------
// Mean pool over T then tiny classifier head (32x2 f32 output).
// ---------------------------------------------------------------------------
__global__ __launch_bounds__(256)
void k_head(const unsigned short* __restrict__ X, const float* __restrict__ Wh,
            const float* __restrict__ bh, float* __restrict__ out) {
  __shared__ float pooled[512];
  const int b = blockIdx.x, tid = threadIdx.x;
  for (int d = tid; d < 512; d += 256) {
    float acc = 0.f;
    for (int t = 0; t < 512; ++t)
      acc += bfu2f(X[((size_t)(b * 512 + t)) * 512 + d]);
    pooled[d] = acc * (1.0f / 512.0f);
  }
  __syncthreads();
  if (tid < 2) {
    float acc = bh[tid];
    for (int d = 0; d < 512; ++d) acc = fmaf(pooled[d], Wh[tid * 512 + d], acc);
    out[b * 2 + tid] = acc;
  }
}

// ---------------------------------------------------------------------------
extern "C" void kernel_launch(void* const* d_in, const int* in_sizes, int n_in,
                              void* d_out, int out_size, void* d_ws, size_t ws_size,
                              hipStream_t stream) {
  (void)in_sizes; (void)n_in; (void)out_size; (void)ws_size;
  const float* x       = (const float*)d_in[0];   // (32,200,512)
  const float* W_embed = (const float*)d_in[1];   // (512,200)
  const float* b_embed = (const float*)d_in[2];   // (512)
  const float* Wproj   = (const float*)d_in[3];   // (4,512,512)
  const float* bproj   = (const float*)d_in[4];   // (4,512)
  const float* fw_re   = (const float*)d_in[5];   // (4,8,64,64,64)
  const float* fw_im   = (const float*)d_in[6];
  const float* W1      = (const float*)d_in[7];   // (4,2048,512)
  const float* W2      = (const float*)d_in[8];   // (4,512,2048)
  const float* Wh      = (const float*)d_in[9];   // (2,512)
  const float* bh      = (const float*)d_in[10];  // (2)
  float* out = (float*)d_out;                     // (32,2)

  char* p = (char*)d_ws;
  size_t off = 0;
  auto take = [&](size_t bytes) -> void* {
    void* r = p + off; off += (bytes + 255) & ~(size_t)255; return r;
  };
  typedef unsigned short u16;
  u16*   X   = (u16*)take(16777216);     // (B,T,D) activations, bf16
  u16*   T1  = (u16*)take(16777216);     // scratch
  u16*   T2  = (u16*)take(16777216);     // scratch
  u16*   F   = (u16*)take(4194304);      // (16384,128) spectrum
  u16*   G   = (u16*)take(4194304);      // (16384,128) mixed
  u16*   Y1  = (u16*)take(67108864);     // (16384,2048) FFN hidden
  u16*   BfT = (u16*)take(131072);       // (128,512) fwd DFT, (N,K)
  u16*   BiT = (u16*)take(131072);       // (512,128) inv DFT, (N,K)
  u16*   WpB = (u16*)take(2097152);      // Wproj bf16
  u16*   W1B = (u16*)take(8388608);      // W1 bf16
  u16*   W2B = (u16*)take(8388608);      // W2 bf16
  float* xT  = (float*)take(13107200);   // (32,512,200) f32
  float* WT  = (float*)take(409600);     // (200,512) f32

  const dim3 tb(32, 8);

  // per-launch prep: DFT tables, weight bf16 conversion, embed
  k_init_dft<<<dim3(256), 256, 0, stream>>>(BfT, BiT);
  k_f2bf<<<dim3(1024), 256, 0, stream>>>((const float4*)Wproj, (ushort4*)WpB, 262144);
  k_f2bf<<<dim3(4096), 256, 0, stream>>>((const float4*)W1, (ushort4*)W1B, 1048576);
  k_f2bf<<<dim3(4096), 256, 0, stream>>>((const float4*)W2, (ushort4*)W2B, 1048576);
  k_transpose_f32<<<dim3(16, 7, 32), tb, 0, stream>>>(x, xT, 200, 512);       // x -> (b,t,n)
  k_transpose_f32<<<dim3(7, 16, 1), tb, 0, stream>>>(W_embed, WT, 512, 200);  // -> (n,d)
  k_embed<<<dim3(16384), 256, 0, stream>>>(xT, WT, b_embed, X);

  for (int l = 0; l < 4; ++l) {
    const u16*   Wp  = WpB + (size_t)l * 262144;
    const float* bp  = bproj + (size_t)l * 512;
    const float* wre = fw_re + (size_t)l * 2097152;
    const float* wim = fw_im + (size_t)l * 2097152;
    const u16*   w1  = W1B + (size_t)l * 1048576;
    const u16*   w2  = W2B + (size_t)l * 1048576;

    // Q = X @ Wproj^T + b                 (16384,512,512)
    k_gemm<true, false, false>
        <<<dim3(8, 128), 256, 0, stream>>>(X, Wp, bp, nullptr, T1, 16384, 512, 512);
    // (b,t,d) -> (b,d,t)
    k_transpose_bf<false><<<dim3(16, 16, 32), tb, 0, stream>>>(T1, nullptr, T2, 512, 512);
    // truncated rFFT: F = Qt @ BfT^T      (16384,128,512)
    k_gemm<false, false, false>
        <<<dim3(2, 128), 256, 0, stream>>>(T2, BfT, nullptr, nullptr, F, 16384, 128, 512);
    // complex per-head mode mixing
    k_modemix<<<dim3(4096), 256, 0, stream>>>(F, wre, wim, G);
    // irFFT: attn(b,d,t) = G @ BiT^T      (16384,512,128)
    k_gemm<false, false, false>
        <<<dim3(8, 128), 256, 0, stream>>>(G, BiT, nullptr, nullptr, T1, 16384, 512, 128);
    // xres(b,t,d) = X + attn^T  -> T2
    k_transpose_bf<true><<<dim3(16, 16, 32), tb, 0, stream>>>(T1, X, T2, 512, 512);
    // Y1 = relu(xres @ W1^T)              (16384,2048,512)
    k_gemm<false, true, false>
        <<<dim3(32, 128), 256, 0, stream>>>(T2, w1, nullptr, nullptr, Y1, 16384, 2048, 512);
    // X = xres + Y1 @ W2^T                (16384,512,2048)
    k_gemm<false, false, true>
        <<<dim3(8, 128), 256, 0, stream>>>(Y1, w2, nullptr, T2, X, 16384, 512, 2048);
  }

  k_head<<<dim3(32), 256, 0, stream>>>(X, Wh, bh, out);
}
